// OriginalIdpSAM_AttentionBlock_20985210208304
// MI455X (gfx1250) — matrix operand order repaired and checked by reading, not compiled, not run on hardware
//
#include <hip/hip_runtime.h>
#include <cstdint>

typedef __bf16 bf16_t;
typedef __attribute__((ext_vector_type(16))) __bf16 v16bf;
typedef __attribute__((ext_vector_type(8)))  float  v8f;

#define LL 512

// ---------------------------------------------------------------- WMMA core
__device__ __forceinline__ v8f wmma_bf16(v16bf a, v16bf b, v8f c) {
  // D = A(16x32 bf16) x B(32x16 bf16) + C(16x16 f32)
  return __builtin_amdgcn_wmma_f32_16x16x32_bf16(false, a, false, b, (short)0, c,
                                                 false, false);
}

__device__ __forceinline__ v8f v8f_zero() {
  v8f z = {0.f, 0.f, 0.f, 0.f, 0.f, 0.f, 0.f, 0.f};
  return z;
}

// A fragment: row-major [16 x >=k0+32] bf16, M = lane&15.
// A k order used here (contiguous; B matches): vgpr v -> K = 2v + 16*half + {0,1}.
__device__ __forceinline__ v16bf load_a_bf(const bf16_t* A, int lda, int k0) {
  const int lane = threadIdx.x & 31;
  const int m = lane & 15, hf = lane >> 4;
  const bf16_t* base = A + m * lda + k0;
  v16bf r;
#pragma unroll
  for (int v = 0; v < 8; ++v) {
    int kb = 2 * v + 16 * hf;
    r[2 * v]     = base[kb];
    r[2 * v + 1] = base[kb + 1];
  }
  return r;
}

// B fragment from weight-style storage W[N][K] (computes A @ W^T). N = lane&15.
// B k order used here (contiguous, same as A): lanes 0-15 hold K=0..15 (2 per vgpr), lanes 16-31 hold K=16..31.
__device__ __forceinline__ v16bf load_bt_f32(const float* W, int ldw, int k0) {
  const int lane = threadIdx.x & 31;
  const int n = lane & 15, hf = lane >> 4;
  const float* base = W + (size_t)n * ldw + k0 + 16 * hf;
  v16bf r;
#pragma unroll
  for (int v = 0; v < 8; ++v) {
    r[2 * v]     = (bf16_t)base[2 * v];
    r[2 * v + 1] = (bf16_t)base[2 * v + 1];
  }
  return r;
}

// Natural B[K][N] from f32 (row-major over K) with K guard (zeros past kmax).
__device__ __forceinline__ v16bf load_b_nat_guard_f32(const float* B, int ldb,
                                                      int kmax) {
  const int lane = threadIdx.x & 31;
  const int n = lane & 15, hf = lane >> 4;
  v16bf r;
#pragma unroll
  for (int v = 0; v < 8; ++v) {
    int k = 2 * v + 16 * hf;
    float x0 = (k < kmax) ? B[(size_t)k * ldb + n] : 0.f;
    float x1 = (k + 1 < kmax) ? B[(size_t)(k + 1) * ldb + n] : 0.f;
    r[2 * v]     = (bf16_t)x0;
    r[2 * v + 1] = (bf16_t)x1;
  }
  return r;
}

// W_e [8 x 64] f32 padded to N=16 with zeros (weight-style B^T fragment).
__device__ __forceinline__ v16bf load_we_frag(const float* We, int k0) {
  const int lane = threadIdx.x & 31;
  const int n = lane & 15, hf = lane >> 4;
  v16bf r;
#pragma unroll
  for (int v = 0; v < 8; ++v) {
    int k = k0 + 2 * v + 16 * hf;
    float a0 = (n < 8) ? We[n * 64 + k] : 0.f;
    float a1 = (n < 8) ? We[n * 64 + k + 1] : 0.f;
    r[2 * v]     = (bf16_t)a0;
    r[2 * v + 1] = (bf16_t)a1;
  }
  return r;
}

// A fragment of the 16x16 prob tile, zero-padded to K=32.
__device__ __forceinline__ v16bf load_probs_frag(const bf16_t* P, int ldp) {
  const int lane = threadIdx.x & 31;
  const int m = lane & 15, hf = lane >> 4;
  v16bf r;
#pragma unroll
  for (int v = 0; v < 8; ++v) {
    int kb = 2 * v + 16 * hf;
    r[2 * v]     = (kb < 16)     ? P[m * ldp + kb]     : (bf16_t)0.f;
    r[2 * v + 1] = (kb + 1 < 16) ? P[m * ldp + kb + 1] : (bf16_t)0.f;
  }
  return r;
}

__device__ __forceinline__ float block_sum_256(float v, float* red) {
#pragma unroll
  for (int o = 16; o >= 1; o >>= 1) v += __shfl_xor(v, o, 32);
  __syncthreads();
  if ((threadIdx.x & 31) == 0) red[threadIdx.x >> 5] = v;
  __syncthreads();
  float t = 0.f;
#pragma unroll
  for (int i = 0; i < 8; ++i) t += red[i];
  return t;
}

// ------------------------------------------------------ generic GEMM kernel
// out[4096 x N] = act(A[4096 x K] @ W[N x K]^T + bias + add0)
__global__ __launch_bounds__(128) void gemm_rowtile(
    const float* __restrict__ A, int K, const float* __restrict__ W,
    const float* __restrict__ bias, const float* __restrict__ add0,
    float* __restrict__ out, int N, int act) {
  __shared__ bf16_t As[16][520];  // K <= 512 (+8 pad)
  const int tid = threadIdx.x;
  const int row0 = blockIdx.x * 16;
  for (int idx = tid; idx < 16 * K; idx += 128) {
    int r = idx / K, c = idx - r * K;
    As[r][c] = (bf16_t)A[(size_t)(row0 + r) * K + c];
  }
  __syncthreads();
  const int wave = tid >> 5, lane = tid & 31;
  const int nl = lane & 15, hf = lane >> 4;
  const int ntiles = N >> 4;
  for (int nt = wave; nt < ntiles; nt += 4) {
    const int n0 = nt * 16;
    v8f acc = v8f_zero();
    for (int k0 = 0; k0 < K; k0 += 32) {
      v16bf a = load_a_bf(&As[0][0], 520, k0);
      v16bf b = load_bt_f32(W + (size_t)n0 * K, K, k0);
      acc = wmma_bf16(a, b, acc);
    }
    const int col = n0 + nl;
    const float bv = bias ? bias[col] : 0.f;
#pragma unroll
    for (int r = 0; r < 8; ++r) {
      const int row = row0 + r + 8 * hf;
      float x = acc[r] + bv;
      if (add0) x += add0[(size_t)row * N + col];
      if (act == 1) x = x / (1.f + __expf(-x));  // silu
      out[(size_t)row * N + col] = x;
    }
  }
}

// ----------------------------------------------- adaLN modulate: ln*(1+sc)+sh
__global__ __launch_bounds__(256) void ln_mod_kernel(
    const float* __restrict__ hin, const float* __restrict__ mvec, int sh_off,
    int sc_off, float* __restrict__ out) {
  __shared__ float red[8];
  const int row = blockIdx.x, c = threadIdx.x;
  float x = hin[(size_t)row * 256 + c];
  float mu = block_sum_256(x, red) * (1.f / 256.f);
  float d = x - mu;
  float var = block_sum_256(d * d, red) * (1.f / 256.f);
  float y = d * rsqrtf(var + 1e-5f);
  const size_t mb = (size_t)row * 1536;
  out[(size_t)row * 256 + c] =
      y * (1.f + mvec[mb + sc_off + c]) + mvec[mb + sh_off + c];
}

// ------------------------------------------------- ln(a+b)*g+bt (affine LN)
__global__ __launch_bounds__(256) void add_ln_affine_kernel(
    const float* __restrict__ a, const float* __restrict__ b,
    const float* __restrict__ g, const float* __restrict__ bt,
    float* __restrict__ out) {
  __shared__ float red[8];
  const int row = blockIdx.x, c = threadIdx.x;
  float x = a[(size_t)row * 256 + c] + b[(size_t)row * 256 + c];
  float mu = block_sum_256(x, red) * (1.f / 256.f);
  float d = x - mu;
  float var = block_sum_256(d * d, red) * (1.f / 256.f);
  out[(size_t)row * 256 + c] = d * rsqrtf(var + 1e-5f) * g[c] + bt[c];
}

// ----------------------------------------------------- out = x*gate + res
__global__ void gate_res_kernel(const float* __restrict__ x,
                                const float* __restrict__ mvec, int off,
                                const float* __restrict__ res,
                                float* __restrict__ out, int total) {
  for (int i = blockIdx.x * blockDim.x + threadIdx.x; i < total;
       i += gridDim.x * blockDim.x) {
    int row = i >> 8, col = i & 255;
    out[i] = x[i] * mvec[(size_t)row * 1536 + off + col] + res[i];
  }
}

// --------------------------------- fused pair-bias + flash attention kernel
// grid (32, 8): x = 16-query tile, y = batch. 256 threads = 8 waves = 8 heads.
// p (512 MB, the bandwidth bottleneck) is streamed via double-buffered
// global_load_async_to_lds_b128 so the next key-tile is in flight while the
// current one is normalized/projected/consumed.
__global__ __launch_bounds__(256) void attn_kernel(
    const float* __restrict__ qkv, const float* __restrict__ p,
    const float* __restrict__ g_eln, const float* __restrict__ b_eln,
    const float* __restrict__ W_e, const float* __restrict__ b_e,
    float* __restrict__ o_out) {
  __shared__ bf16_t Qs[16][264];
  __shared__ float  Praw[2][256][68];  // raw p tile, async-DMA target
  __shared__ bf16_t Pn[256][72];       // ln(p)*g+b tile, pair = lq*16+lk
  __shared__ float  BiasT[256][8];     // bias per (pair, head)
  __shared__ bf16_t Probs[8][16][24];

  const int tid = threadIdx.x;
  const int h = tid >> 5;  // wave == head
  const int lane = tid & 31;
  const int nl = lane & 15, hf = lane >> 4;
  const int b = blockIdx.y;
  const int l0 = blockIdx.x * 16;
  const int rowq = b * LL + l0;
  const float scale = 0.17677669529663687f;  // 1/sqrt(32)

  for (int idx = tid; idx < 16 * 256; idx += 256) {
    int r = idx >> 8, c = idx & 255;
    Qs[r][c] = (bf16_t)(qkv[(size_t)(rowq + r) * 768 + c] * scale);
  }

  // Issue one key-tile of p (256 pairs x 64 f32) as 16 async B128 ops/wave.
  auto issue_p = [&](int buf, int m0) {
    // slot = 16B chunk id; consecutive slots are contiguous in global memory
    // within each query row group (lk is the innermost L dim of p).
#pragma unroll
    for (int j = 0; j < 16; ++j) {
      int slot = (h * 16 + j) * 32 + lane;  // 0..4095
      int pair = slot >> 4;                 // lq*16 + lk
      int ci = slot & 15;                   // 16B chunk within the 256B row
      int lq = pair >> 4, lk = pair & 15;
      const float* gp =
          p + ((size_t)(rowq + lq) * LL + (m0 + lk)) * 64 + ci * 4;
      unsigned long long ga = (unsigned long long)(uintptr_t)gp;
      unsigned lds = (unsigned)(uintptr_t)&Praw[buf][pair][ci * 4];
      asm volatile("global_load_async_to_lds_b128 %0, %1, off"
                   :: "v"(lds), "v"(ga)
                   : "memory");
    }
  };

  float Mrow[8], Srow[8];
  v8f co0 = v8f_zero(), co1 = v8f_zero();
#pragma unroll
  for (int r = 0; r < 8; ++r) { Mrow[r] = -1e30f; Srow[r] = 0.f; }
  const float beh = b_e[h];

  issue_p(0, 0);  // prefetch first tile

  for (int mt = 0; mt < 32; ++mt) {
    const int m0 = mt * 16;
    const int rowk = b * LL + m0;
    const int buf = mt & 1;

    if (mt + 1 < 32) {
      issue_p(buf ^ 1, (mt + 1) * 16);
      // in-order completion: <=16 outstanding -> current tile has landed
      asm volatile("s_wait_asynccnt 0x10" ::: "memory");
    } else {
      asm volatile("s_wait_asynccnt 0x0" ::: "memory");
    }
    __syncthreads();  // all waves' async data visible; prior-iter reads done

    {  // one p-row LayerNorm per thread (256 pairs), from LDS
      const float4* p4 = (const float4*)&Praw[buf][tid][0];
      float sum = 0.f, sq = 0.f;
#pragma unroll
      for (int i = 0; i < 16; ++i) {
        float4 v = p4[i];
        sum += v.x + v.y + v.z + v.w;
        sq += v.x * v.x + v.y * v.y + v.z * v.z + v.w * v.w;
      }
      const float mu = sum * (1.f / 64.f);
      const float rstd = rsqrtf(sq * (1.f / 64.f) - mu * mu + 1e-5f);
#pragma unroll
      for (int i = 0; i < 16; ++i) {
        float4 v = p4[i];
        Pn[tid][4 * i + 0] = (bf16_t)((v.x - mu) * rstd * g_eln[4 * i + 0] + b_eln[4 * i + 0]);
        Pn[tid][4 * i + 1] = (bf16_t)((v.y - mu) * rstd * g_eln[4 * i + 1] + b_eln[4 * i + 1]);
        Pn[tid][4 * i + 2] = (bf16_t)((v.z - mu) * rstd * g_eln[4 * i + 2] + b_eln[4 * i + 2]);
        Pn[tid][4 * i + 3] = (bf16_t)((v.w - mu) * rstd * g_eln[4 * i + 3] + b_eln[4 * i + 3]);
      }
    }
    __syncthreads();

    // bias projection: [256 pairs x 64] @ W_e^T -> BiasT; 2 subtiles per wave
#pragma unroll
    for (int s = 0; s < 2; ++s) {
      const int pair0 = (h * 2 + s) * 16;
      v8f acc = v8f_zero();
      for (int k0 = 0; k0 < 64; k0 += 32) {
        v16bf a = load_a_bf(&Pn[pair0][0], 72, k0);
        v16bf bb = load_we_frag(W_e, k0);
        acc = wmma_bf16(a, bb, acc);
      }
      if (nl < 8) {
#pragma unroll
        for (int r = 0; r < 8; ++r) BiasT[pair0 + r + 8 * hf][nl] = acc[r];
      }
    }
    __syncthreads();

    // scores = (q*scale) @ k^T : k fragment straight from global (L2-hot)
    v8f sc;
    {
      v16bf a = load_a_bf(&Qs[0][h * 32], 264, 0);
      v16bf bb = load_bt_f32(qkv + (size_t)rowk * 768 + 256 + h * 32, 768, 0);
      sc = wmma_bf16(a, bb, v8f_zero());
    }
#pragma unroll
    for (int r = 0; r < 8; ++r) {
      const int q = r + 8 * hf;
      float sv = sc[r] + BiasT[q * 16 + nl][h] + beh;
      float mx = sv;
#pragma unroll
      for (int o = 8; o >= 1; o >>= 1) mx = fmaxf(mx, __shfl_xor(mx, o, 32));
      const float newM = fmaxf(Mrow[r], mx);
      const float corr = __expf(Mrow[r] - newM);
      const float pr = __expf(sv - newM);
      float ts = pr;
#pragma unroll
      for (int o = 8; o >= 1; o >>= 1) ts += __shfl_xor(ts, o, 32);
      Srow[r] = Srow[r] * corr + ts;
      Mrow[r] = newM;
      co0[r] = co0[r] * corr;
      co1[r] = co1[r] * corr;
      Probs[h][q][nl] = (bf16_t)pr;
    }
    // o += probs @ v (K padded 16->32 with zeros); v straight from global
    {
      v16bf ap = load_probs_frag(&Probs[h][0][0], 24);
      v16bf bv0 =
          load_b_nat_guard_f32(qkv + (size_t)rowk * 768 + 512 + h * 32, 768, 16);
      co0 = wmma_bf16(ap, bv0, co0);
      v16bf bv1 = load_b_nat_guard_f32(
          qkv + (size_t)rowk * 768 + 512 + h * 32 + 16, 768, 16);
      co1 = wmma_bf16(ap, bv1, co1);
    }
  }

#pragma unroll
  for (int r = 0; r < 8; ++r) {
    const int q = r + 8 * hf;
    const float inv = 1.f / Srow[r];
    const size_t base = (size_t)(rowq + q) * 256 + h * 32;
    o_out[base + nl] = co0[r] * inv;
    o_out[base + 16 + nl] = co1[r] * inv;
  }
}

// ------------------------------------------------------------------ launch
extern "C" void kernel_launch(void* const* d_in, const int* in_sizes, int n_in,
                              void* d_out, int out_size, void* d_ws,
                              size_t ws_size, hipStream_t stream) {
  (void)in_sizes; (void)n_in; (void)out_size; (void)ws_size;
  const float* h      = (const float*)d_in[0];
  const float* h_init = (const float*)d_in[1];
  const float* s_a    = (const float*)d_in[2];
  const float* s_t    = (const float*)d_in[3];
  const float* s_tem  = (const float*)d_in[4];
  const float* p      = (const float*)d_in[5];
  const float* W_bp   = (const float*)d_in[6];
  const float* b_bp   = (const float*)d_in[7];
  const float* W_ada  = (const float*)d_in[8];
  const float* b_ada  = (const float*)d_in[9];
  const float* g_eln  = (const float*)d_in[10];
  const float* b_eln  = (const float*)d_in[11];
  const float* W_e    = (const float*)d_in[12];
  const float* b_e    = (const float*)d_in[13];
  const float* W_in   = (const float*)d_in[14];
  const float* b_in   = (const float*)d_in[15];
  const float* W_out  = (const float*)d_in[16];
  const float* b_out  = (const float*)d_in[17];
  const float* W1     = (const float*)d_in[18];
  const float* b1     = (const float*)d_in[19];
  const float* W2     = (const float*)d_in[20];
  const float* b2     = (const float*)d_in[21];
  const float* W_tem  = (const float*)d_in[22];
  const float* b_tem  = (const float*)d_in[23];
  const float* g_tem  = (const float*)d_in[24];
  const float* bt_tem = (const float*)d_in[25];
  const float* W_init = (const float*)d_in[26];
  const float* b_init = (const float*)d_in[27];
  const float* g_init = (const float*)d_in[28];
  const float* bt_init= (const float*)d_in[29];

  float* ws = (float*)d_ws;
  const size_t R = 4096;  // B*L
  float* s_buf   = ws; ws += R * 256;
  float* m_buf   = ws; ws += R * 1536;
  float* x_buf   = ws; ws += R * 256;
  float* qkv_buf = ws; ws += R * 768;
  float* o_buf   = ws; ws += R * 256;
  float* o2_buf  = ws; ws += R * 256;
  float* h1_buf  = ws; ws += R * 256;
  float* hid_buf = ws; ws += R * 512;
  float* y_buf   = ws; ws += R * 256;
  float* h2_buf  = ws; ws += R * 256;
  float* t_buf   = ws; ws += R * 256;
  float* h3_buf  = ws; ws += R * 256;

  const dim3 gg(256), gb(128);
  const int TOT = 4096 * 256;

  // conditioning: s = silu(s_a@W_bp^T + b_bp + s_t); m = s@W_ada^T + b_ada
  gemm_rowtile<<<gg, gb, 0, stream>>>(s_a, 64, W_bp, b_bp, s_t, s_buf, 256, 1);
  gemm_rowtile<<<gg, gb, 0, stream>>>(s_buf, 256, W_ada, b_ada, nullptr, m_buf, 1536, 0);
  // attention branch
  ln_mod_kernel<<<4096, 256, 0, stream>>>(h, m_buf, 0, 256, x_buf);
  gemm_rowtile<<<gg, gb, 0, stream>>>(x_buf, 256, W_in, b_in, nullptr, qkv_buf, 768, 0);
  attn_kernel<<<dim3(32, 8), 256, 0, stream>>>(qkv_buf, p, g_eln, b_eln, W_e, b_e, o_buf);
  gemm_rowtile<<<gg, gb, 0, stream>>>(o_buf, 256, W_out, b_out, nullptr, o2_buf, 256, 0);
  gate_res_kernel<<<1024, 256, 0, stream>>>(o2_buf, m_buf, 512, h, h1_buf, TOT);
  // MLP branch
  ln_mod_kernel<<<4096, 256, 0, stream>>>(h1_buf, m_buf, 768, 1024, x_buf);
  gemm_rowtile<<<gg, gb, 0, stream>>>(x_buf, 256, W1, b1, nullptr, hid_buf, 512, 1);
  gemm_rowtile<<<gg, gb, 0, stream>>>(hid_buf, 512, W2, b2, nullptr, y_buf, 256, 0);
  gate_res_kernel<<<1024, 256, 0, stream>>>(y_buf, m_buf, 1280, h1_buf, h2_buf, TOT);
  // trailing conditioning LayerNorms
  gemm_rowtile<<<gg, gb, 0, stream>>>(s_tem, 128, W_tem, b_tem, nullptr, t_buf, 256, 0);
  add_ln_affine_kernel<<<4096, 256, 0, stream>>>(h2_buf, t_buf, g_tem, bt_tem, h3_buf);
  gemm_rowtile<<<gg, gb, 0, stream>>>(h_init, 256, W_init, b_init, nullptr, t_buf, 256, 0);
  add_ln_affine_kernel<<<4096, 256, 0, stream>>>(h3_buf, t_buf, g_init, bt_init, (float*)d_out);
}
